// AutoGraphModel_GAT_17514876633767
// MI455X (gfx1250) — compile-verified, matrix operand order checked by reading, not run on hardware
//
#include <hip/hip_runtime.h>
#include <hip/hip_bf16.h>

// ---------------------------------------------------------------------------
// AutoGraphModel_GAT for MI455X (gfx1250, wave32)
//  - message passing: L2-resident gather + float atomics (memory/atomic bound)
//  - x1 @ W2 (100k x 128 x 128): f16 WMMA (v_wmma_f32_16x16x32_f16)
//    * W2^T staged once per block in LDS (32 KB), read as ds_load_b128
//    * each wave owns a 16-row tile; A fragments register-resident, reused
//      across all 8 column tiles -> 32 WMMAs per wave, 1x A traffic
// ---------------------------------------------------------------------------

#define NNODES 100000

typedef _Float16 v16h __attribute__((ext_vector_type(16)));
typedef float    v8f  __attribute__((ext_vector_type(8)));

// ---- float <-> order-preserving uint (for float atomicMax) ----
__device__ __forceinline__ unsigned f2ord(float f) {
  unsigned u = __float_as_uint(f);
  return (u & 0x80000000u) ? ~u : (u | 0x80000000u);
}
__device__ __forceinline__ float ord2f(unsigned u) {
  return (u & 0x80000000u) ? __uint_as_float(u & 0x7fffffffu)
                           : __uint_as_float(~u);
}

// ---------------- utility fills ----------------
__global__ void fill_f32(float* p, float v, int n) {
  int i = blockIdx.x * blockDim.x + threadIdx.x;
  if (i < n) p[i] = v;
}
__global__ void fill_u32(unsigned* p, unsigned v, int n) {
  int i = blockIdx.x * blockDim.x + threadIdx.x;
  if (i < n) p[i] = v;
}

// ---------------- degrees (self-loops pre-counted by init=1.0) -------------
__global__ void degrees_k(const int* __restrict__ src, const int* __restrict__ dst,
                          float* __restrict__ hdeg, int E) {
  int e = blockIdx.x * blockDim.x + threadIdx.x;
  if (e >= E) return;
  atomicAdd(&hdeg[dst[e] * 2 + 0], 1.0f);  // indeg
  atomicAdd(&hdeg[src[e] * 2 + 1], 1.0f);  // outdeg
}

// ---------------- GATConv1 node phase: feat1 = h@W1, el1/er1 ---------------
__global__ void node1_k(const float* __restrict__ hdeg, const float* __restrict__ W1,
                        const float* __restrict__ al1, const float* __restrict__ ar1,
                        float* __restrict__ feat, float* __restrict__ el,
                        float* __restrict__ er) {
  int n = blockIdx.x;
  int t = threadIdx.x;  // 128
  float h0 = hdeg[n * 2 + 0], h1v = hdeg[n * 2 + 1];
  float f = h0 * W1[t] + h1v * W1[128 + t];
  feat[n * 128 + t] = f;
  int head = t >> 6, d = t & 63;
  __shared__ float sl[128], sr[128];
  sl[t] = f * al1[head * 64 + d];
  sr[t] = f * ar1[head * 64 + d];
  __syncthreads();
  for (int off = 32; off >= 1; off >>= 1) {
    if ((t & 63) < off) { sl[t] += sl[t + off]; sr[t] += sr[t + off]; }
    __syncthreads();
  }
  if (t == 0) {
    el[n * 2 + 0] = sl[0];  el[n * 2 + 1] = sl[64];
    er[n * 2 + 0] = sr[0];  er[n * 2 + 1] = sr[64];
  }
}

// ---------------- GATConv2 node phase scores (H=1, D=128) ------------------
__global__ void node2_k(const float* __restrict__ feat, const float* __restrict__ al2,
                        const float* __restrict__ ar2, float* __restrict__ el,
                        float* __restrict__ er) {
  int n = blockIdx.x;
  int t = threadIdx.x;  // 128
  float f = feat[n * 128 + t];
  __shared__ float sl[128], sr[128];
  sl[t] = f * al2[t];
  sr[t] = f * ar2[t];
  __syncthreads();
  for (int off = 64; off >= 1; off >>= 1) {
    if (t < off) { sl[t] += sl[t + off]; sr[t] += sr[t + off]; }
    __syncthreads();
  }
  if (t == 0) { el[n] = sl[0]; er[n] = sr[0]; }
}

// ---------------- edge passes ----------------------------------------------
template <int H>
__global__ void edge_max_k(const int* __restrict__ src, const int* __restrict__ dst,
                           const float* __restrict__ el, const float* __restrict__ er,
                           unsigned* __restrict__ mo, int E, int E2) {
  int e = blockIdx.x * blockDim.x + threadIdx.x;
  if (e >= E2) return;
  int s = (e < E) ? src[e] : (e - E);
  int d = (e < E) ? dst[e] : (e - E);
#pragma unroll
  for (int h = 0; h < H; ++h) {
    float v = el[s * H + h] + er[d * H + h];
    v = v < 0.f ? 0.2f * v : v;  // LeakyReLU(0.2)
    atomicMax(mo + d * H + h, f2ord(v));
  }
}

template <int H>
__global__ void edge_sum_k(const int* __restrict__ src, const int* __restrict__ dst,
                           const float* __restrict__ el, const float* __restrict__ er,
                           const unsigned* __restrict__ mo, float* __restrict__ z,
                           int E, int E2) {
  int e = blockIdx.x * blockDim.x + threadIdx.x;
  if (e >= E2) return;
  int s = (e < E) ? src[e] : (e - E);
  int d = (e < E) ? dst[e] : (e - E);
#pragma unroll
  for (int h = 0; h < H; ++h) {
    float v = el[s * H + h] + er[d * H + h];
    v = v < 0.f ? 0.2f * v : v;
    atomicAdd(z + d * H + h, expf(v - ord2f(mo[d * H + h])));
  }
}

// one wave (32 lanes) per edge: 4 features per lane, 4 atomic f32 adds
template <int H, int D>
__global__ void edge_scatter_k(const int* __restrict__ src, const int* __restrict__ dst,
                               const float* __restrict__ el, const float* __restrict__ er,
                               const unsigned* __restrict__ mo, const float* __restrict__ z,
                               const float* __restrict__ feat, float* __restrict__ acc,
                               int E, int E2) {
  int wid = blockIdx.x * (blockDim.x >> 5) + (threadIdx.x >> 5);
  if (wid >= E2) return;
  int lane = threadIdx.x & 31;
  int s = (wid < E) ? src[wid] : (wid - E);
  int d = (wid < E) ? dst[wid] : (wid - E);
  int f0 = lane * 4;
  int h = f0 / D;  // compile-time D -> head is uniform over the float4
  float v = el[s * H + h] + er[d * H + h];
  v = v < 0.f ? 0.2f * v : v;
  float alpha = expf(v - ord2f(mo[d * H + h])) / z[d * H + h];
  const float4 fv = *(const float4*)(feat + s * 128 + f0);
  float* ap = acc + d * 128 + f0;
  atomicAdd(ap + 0, fv.x * alpha);
  atomicAdd(ap + 1, fv.y * alpha);
  atomicAdd(ap + 2, fv.z * alpha);
  atomicAdd(ap + 3, fv.w * alpha);
}

// ---------------- finalize (bias + ELU) ------------------------------------
__global__ void finalize_h16_k(const float* __restrict__ acc, const float* __restrict__ b,
                               _Float16* __restrict__ out, int total) {
  int i = blockIdx.x * blockDim.x + threadIdx.x;
  if (i >= total) return;
  float x = acc[i] + b[i & 127];
  x = x > 0.f ? x : (expf(x) - 1.f);
  out[i] = (_Float16)x;
}
__global__ void finalize_f32_k(const float* __restrict__ acc, const float* __restrict__ b,
                               float* __restrict__ out, int total) {
  int i = blockIdx.x * blockDim.x + threadIdx.x;
  if (i >= total) return;
  float x = acc[i] + b[i & 127];
  x = x > 0.f ? x : (expf(x) - 1.f);
  out[i] = x;
}

// ---------------- W2 -> f16 transposed -------------------------------------
__global__ void convert_w2_k(const float* __restrict__ W2, _Float16* __restrict__ w2t) {
  int i = blockIdx.x * blockDim.x + threadIdx.x;  // i = n*128 + k
  if (i >= 128 * 128) return;
  int n = i >> 7, k = i & 127;
  w2t[i] = (_Float16)W2[k * 128 + n];
}

// ---------------- WMMA GEMM: C[N,128] = A[N,128] * W2[128,128] -------------
// A f16 row-major, Bt = W2^T f16 (Bt[n,k] = W2[k,n]) staged in LDS.
// Each wave owns one 16-row tile (8 tiles per block); the 4 K-step A
// fragments are loaded once into registers and reused across all 8 column
// tiles -> 32 v_wmma per wave, A global traffic 1x.
__global__ void __launch_bounds__(256)
gemm_wmma_k(const _Float16* __restrict__ A, const _Float16* __restrict__ Bt,
            float* __restrict__ C, int nrows) {
  __shared__ _Float16 ldsB[128 * 128];  // 32 KB
  const int tid = threadIdx.x;
  {
    const uint4* s = (const uint4*)Bt;
    uint4* d = (uint4*)ldsB;
    for (int i = tid; i < (128 * 128) / 8; i += 256) d[i] = s[i];
  }
  __syncthreads();

  const int wave = tid >> 5;  // wave32: 8 waves per block
  const int lane = tid & 31;
  const int r = lane & 15;            // A row in tile / B col in tile
  const int kgrp = (lane >> 4) * 8;   // ISA 16-bit fragment K grouping

  const int row0 = (blockIdx.x * 8 + wave) * 16;
  if (row0 >= nrows) return;  // after the only __syncthreads -> safe

  union F { v16h v; uint4 u[2]; };

  // A fragments: register-resident, one per K-step
  const _Float16* Arow = A + (size_t)(row0 + r) * 128;
  F a[4];
#pragma unroll
  for (int ks = 0; ks < 4; ++ks) {
    const int kb = ks * 32 + kgrp;
    a[ks].u[0] = *(const uint4*)(Arow + kb);        // K kb..kb+7
    a[ks].u[1] = *(const uint4*)(Arow + kb + 16);   // K kb+16..kb+23
  }

  const int col = lane & 15;
  const int rbase = (lane >> 4) * 8;  // C/D layout rows
#pragma unroll
  for (int ct = 0; ct < 8; ++ct) {    // sweep the 8 output column tiles
    const _Float16* Bcol = ldsB + (ct * 16 + r) * 128;
    v8f c = {};
#pragma unroll
    for (int ks = 0; ks < 4; ++ks) {
      const int kb = ks * 32 + kgrp;
      F b;
      b.u[0] = *(const uint4*)(Bcol + kb);
      b.u[1] = *(const uint4*)(Bcol + kb + 16);
      c = __builtin_amdgcn_wmma_f32_16x16x32_f16(false, a[ks].v, false, b.v,
                                                 (short)0, c, false, false);
    }
    float* Cp = C + (size_t)(row0 + rbase) * 128 + ct * 16 + col;
#pragma unroll
    for (int i = 0; i < 8; ++i) Cp[i * 128] = c[i];
  }
}

// ---------------- column mean over nodes -----------------------------------
__global__ void col_mean_k(const float* __restrict__ h, float* __restrict__ hg, int n) {
  int f = blockIdx.x;   // 128 blocks
  int t = threadIdx.x;  // 256
  float s = 0.f;
  for (int i = t; i < n; i += 256) s += h[i * 128 + f];
  __shared__ float red[256];
  red[t] = s;
  __syncthreads();
  for (int off = 128; off >= 1; off >>= 1) {
    if (t < off) red[t] += red[t + off];
    __syncthreads();
  }
  if (t == 0) hg[f] = red[0] / (float)n;
}

// ---------------- head MLPs (single block) ---------------------------------
__global__ void head_mlp_k(const float* __restrict__ hg,
                           const float* __restrict__ fc1_w, const float* __restrict__ fc1_b,
                           const int* __restrict__ sched,
                           const float* __restrict__ e_dir, const float* __restrict__ e_par,
                           const float* __restrict__ e_fr, const float* __restrict__ e_ssg,
                           const float* __restrict__ sw1, const float* __restrict__ sb1,
                           const float* __restrict__ sw2, const float* __restrict__ sb2,
                           const float* __restrict__ fw1, const float* __restrict__ fb1,
                           const float* __restrict__ fw2, const float* __restrict__ fb2,
                           const float* __restrict__ fw3, const float* __restrict__ fb3,
                           float* __restrict__ out) {
  __shared__ float hgl[128], xy[256], tmp[128], hid[128], red[64];
  int t = threadIdx.x;  // 256
  if (t < 128) hgl[t] = hg[t];
  if (t < 128) {  // schedule embedding -> tmp
    int seg = t >> 5, off = t & 31;
    float v;
    if (seg == 0)      v = e_dir[sched[0] * 32 + off];
    else if (seg == 1) v = e_par[sched[1] * 32 + off];
    else if (seg == 2) v = e_fr [sched[2] * 32 + off];
    else               v = e_ssg[sched[3] * 32 + off];
    tmp[t] = v;
  }
  __syncthreads();
  if (t < 128) {  // gfeat = hg @ fc1_w + fc1_b
    float a = fc1_b[t];
    for (int k = 0; k < 128; ++k) a += hgl[k] * fc1_w[k * 128 + t];
    xy[t] = a;
  }
  if (t < 128) {  // relu(se @ sw1 + sb1)
    float a = sb1[t];
    for (int k = 0; k < 128; ++k) a += tmp[k] * sw1[k * 128 + t];
    hid[t] = a > 0.f ? a : 0.f;
  }
  __syncthreads();
  if (t < 128) {  // sfeat = hid @ sw2 + sb2
    float a = sb2[t];
    for (int k = 0; k < 128; ++k) a += hid[k] * sw2[k * 128 + t];
    xy[128 + t] = a;
  }
  __syncthreads();
  if (t < 128) {  // h1 = relu(xy @ fw1 + fb1)
    float a = fb1[t];
    for (int k = 0; k < 256; ++k) a += xy[k] * fw1[k * 128 + t];
    hid[t] = a > 0.f ? a : 0.f;
  }
  __syncthreads();
  if (t < 64) {  // h2 = relu(h1 @ fw2 + fb2); dot with fw3
    float a = fb2[t];
    for (int k = 0; k < 128; ++k) a += hid[k] * fw2[k * 64 + t];
    a = a > 0.f ? a : 0.f;
    red[t] = a * fw3[t];
  }
  __syncthreads();
  for (int off = 32; off >= 1; off >>= 1) {
    if (t < off) red[t] += red[t + off];
    __syncthreads();
  }
  if (t == 0) out[0] = red[0] + fb3[0];
}

// ---------------------------------------------------------------------------
extern "C" void kernel_launch(void* const* d_in, const int* in_sizes, int n_in,
                              void* d_out, int out_size, void* d_ws, size_t ws_size,
                              hipStream_t stream) {
  const int N  = NNODES;
  const int E  = in_sizes[0];
  const int E2 = E + N;

  const int*   src   = (const int*)d_in[0];
  const int*   dst   = (const int*)d_in[1];
  const int*   sched = (const int*)d_in[2];
  const float* W1    = (const float*)d_in[3];
  const float* al1   = (const float*)d_in[4];
  const float* ar1   = (const float*)d_in[5];
  const float* b1    = (const float*)d_in[6];
  const float* W2    = (const float*)d_in[7];
  const float* al2   = (const float*)d_in[8];
  const float* ar2   = (const float*)d_in[9];
  const float* b2    = (const float*)d_in[10];
  const float* fc1_w = (const float*)d_in[11];
  const float* fc1_b = (const float*)d_in[12];
  const float* e_dir = (const float*)d_in[13];
  const float* e_par = (const float*)d_in[14];
  const float* e_fr  = (const float*)d_in[15];
  const float* e_ssg = (const float*)d_in[16];
  const float* sw1   = (const float*)d_in[17];
  const float* sb1   = (const float*)d_in[18];
  const float* sw2   = (const float*)d_in[19];
  const float* sb2   = (const float*)d_in[20];
  const float* fw1   = (const float*)d_in[21];
  const float* fb1   = (const float*)d_in[22];
  const float* fw2   = (const float*)d_in[23];
  const float* fb2   = (const float*)d_in[24];
  const float* fw3   = (const float*)d_in[25];
  const float* fb3   = (const float*)d_in[26];

  // ---- workspace carve-out ----
  char* w = (char*)d_ws;
  auto carve = [&](size_t bytes) -> void* {
    void* p = (void*)w;
    w += (bytes + 255) & ~((size_t)255);
    return p;
  };
  float*    featbuf = (float*)carve(sizeof(float) * (size_t)N * 128);  // feat1->feat2->h2
  float*    accbuf  = (float*)carve(sizeof(float) * (size_t)N * 128);  // acc1->acc2
  _Float16* x1h     = (_Float16*)carve(sizeof(_Float16) * (size_t)N * 128);
  float*    hdeg    = (float*)carve(sizeof(float) * (size_t)N * 2);
  float*    el1     = (float*)carve(sizeof(float) * (size_t)N * 2);
  float*    er1     = (float*)carve(sizeof(float) * (size_t)N * 2);
  unsigned* m1      = (unsigned*)carve(sizeof(unsigned) * (size_t)N * 2);
  float*    z1      = (float*)carve(sizeof(float) * (size_t)N * 2);
  float*    el2     = (float*)carve(sizeof(float) * (size_t)N);
  float*    er2     = (float*)carve(sizeof(float) * (size_t)N);
  unsigned* m2      = (unsigned*)carve(sizeof(unsigned) * (size_t)N);
  float*    z2      = (float*)carve(sizeof(float) * (size_t)N);
  _Float16* w2t     = (_Float16*)carve(sizeof(_Float16) * 128 * 128);
  float*    hg      = (float*)carve(sizeof(float) * 128);

  const int B = 256;
  auto g = [&](int n) { return (n + B - 1) / B; };
  const int NF = N * 128;

  // ---- layer 1 init (self-loops contribute 1 to each degree) ----
  fill_f32<<<g(2 * N), B, 0, stream>>>(hdeg, 1.0f, 2 * N);
  fill_u32<<<g(2 * N), B, 0, stream>>>(m1, 0u, 2 * N);  // 0 < ord(-inf) < all finite
  fill_f32<<<g(2 * N), B, 0, stream>>>(z1, 0.0f, 2 * N);
  fill_f32<<<g(NF), B, 0, stream>>>(accbuf, 0.0f, NF);

  degrees_k<<<g(E), B, 0, stream>>>(src, dst, hdeg, E);
  node1_k<<<N, 128, 0, stream>>>(hdeg, W1, al1, ar1, featbuf, el1, er1);
  edge_max_k<2><<<g(E2), B, 0, stream>>>(src, dst, el1, er1, m1, E, E2);
  edge_sum_k<2><<<g(E2), B, 0, stream>>>(src, dst, el1, er1, m1, z1, E, E2);
  edge_scatter_k<2, 64><<<(E2 + 7) / 8, B, 0, stream>>>(src, dst, el1, er1, m1, z1,
                                                        featbuf, accbuf, E, E2);
  finalize_h16_k<<<g(NF), B, 0, stream>>>(accbuf, b1, x1h, NF);  // x1 = elu(.+b1), f16

  // ---- layer 2: WMMA GEMM feat2 = x1 @ W2 ----
  convert_w2_k<<<g(128 * 128), B, 0, stream>>>(W2, w2t);
  const int rowTiles = (N + 15) / 16;                 // 6250
  gemm_wmma_k<<<(rowTiles + 7) / 8, 256, 0, stream>>>(x1h, w2t, featbuf, N);
  node2_k<<<N, 128, 0, stream>>>(featbuf, al2, ar2, el2, er2);

  fill_u32<<<g(N), B, 0, stream>>>(m2, 0u, N);
  fill_f32<<<g(N), B, 0, stream>>>(z2, 0.0f, N);
  fill_f32<<<g(NF), B, 0, stream>>>(accbuf, 0.0f, NF);

  edge_max_k<1><<<g(E2), B, 0, stream>>>(src, dst, el2, er2, m2, E, E2);
  edge_sum_k<1><<<g(E2), B, 0, stream>>>(src, dst, el2, er2, m2, z2, E, E2);
  edge_scatter_k<1, 128><<<(E2 + 7) / 8, B, 0, stream>>>(src, dst, el2, er2, m2, z2,
                                                         featbuf, accbuf, E, E2);
  finalize_f32_k<<<g(NF), B, 0, stream>>>(accbuf, b2, featbuf, NF);  // h2 in featbuf

  // ---- readout + heads ----
  col_mean_k<<<128, 256, 0, stream>>>(featbuf, hg, N);
  head_mlp_k<<<1, 256, 0, stream>>>(hg, fc1_w, fc1_b, sched, e_dir, e_par, e_fr, e_ssg,
                                    sw1, sb1, sw2, sb2, fw1, fb1, fw2, fb2, fw3, fb3,
                                    (float*)d_out);
}
